// KeypointMHA_1391569404344
// MI455X (gfx1250) — matrix operand order repaired and checked by reading, not compiled, not run on hardware
//
#include <hip/hip_runtime.h>

typedef __attribute__((ext_vector_type(16))) __bf16 v16bf;
typedef __attribute__((ext_vector_type(8)))  __bf16 v8bf;
typedef __attribute__((ext_vector_type(8)))  float  v8f;

// ---------------------------------------------------------------------------
// WMMA fragment helpers (CDNA5 16x16x32 bf16 layouts, wave32)
//   A (16x32): lane l -> row = l&15; elems 0..7 = K[8*kh .. 8*kh+7],
//              elems 8..15 = K[16+8*kh .. 16+8*kh+7]   (kh = l>>4)
//   B (32x16): lane l -> col = l&15; elems 0..7 = K[8*kh .. 8*kh+7],
//              elems 8..15 = K[16+8*kh .. 16+8*kh+7]
// Source tiles are stored K-contiguous per row/column.
// ---------------------------------------------------------------------------
__device__ __forceinline__ v16bf frag_a(const __bf16* rowbase, int kb, int kh) {
  const __bf16* lo = rowbase + kb + 8 * kh;
  v8bf l = *(const v8bf*)lo;
  v8bf h = *(const v8bf*)(lo + 16);
  v16bf r;
#pragma unroll
  for (int i = 0; i < 8; ++i) { r[i] = l[i]; r[i + 8] = h[i]; }
  return r;
}

__device__ __forceinline__ v16bf frag_b(const __bf16* colbase, int kb, int kh) {
  const __bf16* lo = colbase + kb + 8 * kh;
  v8bf l = *(const v8bf*)lo;
  v8bf h = *(const v8bf*)(lo + 16);
  v16bf r;
#pragma unroll
  for (int i = 0; i < 8; ++i) { r[i] = l[i]; r[i + 8] = h[i]; }
  return r;
}

__device__ __forceinline__ v8f wmma_bf16(v16bf a, v16bf b, v8f c) {
  return __builtin_amdgcn_wmma_f32_16x16x32_bf16(false, a, false, b,
                                                 (short)0, c, false, false);
}

__device__ __forceinline__ float silu_f(float x) {
  return x / (1.0f + __expf(-x));
}

// Async global -> LDS copy of 16B per lane (ASYNCcnt-tracked, bypasses VGPRs).
__device__ __forceinline__ void async_copy_b128(const void* lds_dst,
                                                const void* gsrc) {
  unsigned lds = (unsigned)(uintptr_t)lds_dst;            // LDS byte offset
  unsigned long long ga = (unsigned long long)(uintptr_t)gsrc;
  asm volatile("global_load_async_to_lds_b128 %0, %1, off"
               :: "v"(lds), "v"(ga) : "memory");
}

__device__ __forceinline__ void async_wait0() {
  asm volatile("s_wait_asynccnt 0" ::: "memory");
}

// ---------------------------------------------------------------------------
// fp32 -> bf16 conversion
// ---------------------------------------------------------------------------
__global__ __launch_bounds__(256) void k_convert(const float* __restrict__ in,
                                                 __bf16* __restrict__ out, int n) {
  int i = blockIdx.x * 256 + threadIdx.x;
  if (i < n) out[i] = (__bf16)in[i];
}

// ---------------------------------------------------------------------------
// QKV projection GEMM: X[16384,256]bf16 @ W[256,512]bf16 -> Out as [B,H,N,64]
// block = 8 waves, tile M=128 (16 rows/wave), N=64, K-step 32.
// ---------------------------------------------------------------------------
__global__ __launch_bounds__(256) void k_gemm_qkv(const __bf16* __restrict__ X,
                                                  const __bf16* __restrict__ W,
                                                  __bf16* __restrict__ Out) {
  __shared__ __align__(16) __bf16 As[128 * 32];
  __shared__ __align__(16) __bf16 Bs[64 * 32];   // Bt[n][k]
  const int tid = threadIdx.x;
  const int wave = tid >> 5, lane = tid & 31;
  const int r = lane & 15, kh = lane >> 4;
  const int m0 = blockIdx.x * 128, n0 = blockIdx.y * 64;

  v8f acc[4];
#pragma unroll
  for (int i = 0; i < 4; ++i) acc[i] = 0.0f;

  for (int k0 = 0; k0 < 256; k0 += 32) {
    { // stage A tile 128x32 via async global->LDS (2 x 16B per thread)
      int row = tid >> 1, col = (tid & 1) * 16;
      const __bf16* src = &X[(size_t)(m0 + row) * 256 + k0 + col];
      async_copy_b128(&As[row * 32 + col], src);
      async_copy_b128(&As[row * 32 + col + 8], src + 8);
    }
    { // stage Bt tile 64x32 (transpose W[k][n] -> Bt[n][k])
      int k = tid >> 3, n = (tid & 7) * 8;
      v8bf w = *(const v8bf*)&W[(size_t)(k0 + k) * 512 + n0 + n];
#pragma unroll
      for (int j = 0; j < 8; ++j) Bs[(n + j) * 32 + k] = w[j];
    }
    async_wait0();
    __syncthreads();

    v16bf a = frag_a(&As[(wave * 16 + r) * 32], 0, kh);
    v16bf bf4[4];
#pragma unroll
    for (int nt = 0; nt < 4; ++nt)
      bf4[nt] = frag_b(&Bs[(nt * 16 + r) * 32], 0, kh);
#pragma unroll
    for (int nt = 0; nt < 4; ++nt)
      acc[nt] = wmma_bf16(a, bf4[nt], acc[nt]);
    __syncthreads();
  }

  // scatter to [B, H, N, 64] bf16
#pragma unroll
  for (int nt = 0; nt < 4; ++nt)
#pragma unroll
    for (int vr = 0; vr < 8; ++vr) {
      int row = m0 + wave * 16 + vr + 8 * kh;   // b*512 + n_kp
      int col = n0 + nt * 16 + r;               // h*64 + d
      int bb = row >> 9, nk = row & 511;
      int hh = col >> 6, d = col & 63;
      Out[(size_t)((bb * 8 + hh) * 512 + nk) * 64 + d] = (__bf16)acc[nt][vr];
    }
}

// ---------------------------------------------------------------------------
// Fused flash attention per (b, h, 128-query tile). 8 waves x 16 q-rows.
// Score scale: att = dist*sqrt(2) + kq/(8*sqrt(2)); online softmax; P·V fused.
// ---------------------------------------------------------------------------
__global__ __launch_bounds__(256) void k_attn(const float* __restrict__ pos,
                                              const __bf16* __restrict__ Q,
                                              const __bf16* __restrict__ K,
                                              const __bf16* __restrict__ V,
                                              __bf16* __restrict__ upd) {
  __shared__ __align__(16) __bf16 Kt[64 * 64];        // [key][d]
  __shared__ __align__(16) __bf16 Vt[64 * 64];        // [d][key]
  __shared__ __align__(16) __bf16 Pb[8 * 16 * 64];    // per-wave P transpose buf

  const int tid = threadIdx.x;
  const int wave = tid >> 5, lane = tid & 31;
  const int r = lane & 15, kh = lane >> 4;
  const int b = blockIdx.z, h = blockIdx.y;
  const int qw = blockIdx.x * 128 + wave * 16;

  const size_t headbase = (size_t)((b * 8 + h) * 512) * 64;
  const __bf16* Qp = Q + headbase;
  const __bf16* Kp = K + headbase;
  const __bf16* Vp = V + headbase;

  // Q fragments: 16 rows x 64 d = 2 K-steps, resident for whole kernel
  v16bf qf[2];
#pragma unroll
  for (int kk = 0; kk < 2; ++kk)
    qf[kk] = frag_a(Qp + (size_t)(qw + r) * 64, kk * 32, kh);

  // per-lane query positions for its 8 accumulator rows (M = vr + 8*kh)
  float qpx[8], qpy[8], qpz[8];
#pragma unroll
  for (int vr = 0; vr < 8; ++vr) {
    const float* p = pos + (size_t)(b * 512 + qw + vr + 8 * kh) * 3;
    qpx[vr] = p[0]; qpy[vr] = p[1]; qpz[vr] = p[2];
  }

  float mrow[8], lrow[8];
#pragma unroll
  for (int i = 0; i < 8; ++i) { mrow[i] = -1e30f; lrow[i] = 0.0f; }
  v8f o[4];
#pragma unroll
  for (int i = 0; i < 4; ++i) o[i] = 0.0f;

  __bf16* Pw = Pb + wave * (16 * 64);
  const float kq_scale   = 0.0883883476f;  // 1/(8*sqrt(2))
  const float dist_scale = 1.4142135624f;  // sqrt(2)

  for (int k0 = 0; k0 < 512; k0 += 64) {
    { // K tile: pure copy -> async global->LDS; V tile: VALU transpose
      int row = tid >> 2, col = (tid & 3) * 16;
      const __bf16* ks = &Kp[(size_t)(k0 + row) * 64 + col];
      async_copy_b128(&Kt[row * 64 + col], ks);
      async_copy_b128(&Kt[row * 64 + col + 8], ks + 8);
      v8bf v0 = *(const v8bf*)&Vp[(size_t)(k0 + row) * 64 + col];
      v8bf v1 = *(const v8bf*)&Vp[(size_t)(k0 + row) * 64 + col + 8];
#pragma unroll
      for (int j = 0; j < 8; ++j) {
        Vt[(col + j) * 64 + row]     = v0[j];
        Vt[(col + 8 + j) * 64 + row] = v1[j];
      }
    }
    async_wait0();
    __syncthreads();

    // S = Q · K^T  (16 x 64 tile per wave): batch B-frag loads per K-step
    v8f s[4];
#pragma unroll
    for (int nt = 0; nt < 4; ++nt) s[nt] = 0.0f;
#pragma unroll
    for (int kk = 0; kk < 2; ++kk) {
      v16bf kf[4];
#pragma unroll
      for (int nt = 0; nt < 4; ++nt)
        kf[nt] = frag_b(&Kt[(nt * 16 + r) * 64], kk * 32, kh);
#pragma unroll
      for (int nt = 0; nt < 4; ++nt)
        s[nt] = wmma_bf16(qf[kk], kf[nt], s[nt]);
    }

    // add distance term + scaling (lane column = key, vgpr row = query)
#pragma unroll
    for (int nt = 0; nt < 4; ++nt) {
      const float* p = pos + (size_t)(b * 512 + k0 + nt * 16 + r) * 3;
      float kx = p[0], ky = p[1], kz = p[2];
#pragma unroll
      for (int vr = 0; vr < 8; ++vr) {
        float dx = qpx[vr] - kx, dy = qpy[vr] - ky, dz = qpz[vr] - kz;
        float d2 = dx * dx + dy * dy + dz * dz;
        float dist = d2 > 0.0f ? sqrtf(d2) : 0.0f;
        s[nt][vr] = s[nt][vr] * kq_scale + dist * dist_scale;
      }
    }

    // online softmax: row reductions over 16-lane column groups
#pragma unroll
    for (int vr = 0; vr < 8; ++vr) {
      float mx = fmaxf(fmaxf(s[0][vr], s[1][vr]), fmaxf(s[2][vr], s[3][vr]));
#pragma unroll
      for (int msk = 1; msk < 16; msk <<= 1) mx = fmaxf(mx, __shfl_xor(mx, msk, 32));
      float mnew = fmaxf(mrow[vr], mx);
      float scale = __expf(mrow[vr] - mnew);
      mrow[vr] = mnew;
      float psum = 0.0f;
#pragma unroll
      for (int nt = 0; nt < 4; ++nt) {
        float pe = __expf(s[nt][vr] - mnew);
        s[nt][vr] = pe;
        psum += pe;
      }
#pragma unroll
      for (int msk = 1; msk < 16; msk <<= 1) psum += __shfl_xor(psum, msk, 32);
      lrow[vr] = lrow[vr] * scale + psum;
#pragma unroll
      for (int dt = 0; dt < 4; ++dt) o[dt][vr] *= scale;
    }

    // P (C-layout) -> per-wave LDS row-major [m][key], then consume as A-frags
#pragma unroll
    for (int nt = 0; nt < 4; ++nt)
#pragma unroll
      for (int vr = 0; vr < 8; ++vr)
        Pw[(vr + 8 * kh) * 64 + nt * 16 + r] = (__bf16)s[nt][vr];

#pragma unroll
    for (int kk = 0; kk < 2; ++kk) {
      v16bf pf = frag_a(&Pw[r * 64], kk * 32, kh);
      v16bf vf[4];
#pragma unroll
      for (int dt = 0; dt < 4; ++dt)
        vf[dt] = frag_b(&Vt[(dt * 16 + r) * 64], kk * 32, kh);
#pragma unroll
      for (int dt = 0; dt < 4; ++dt)
        o[dt] = wmma_bf16(pf, vf[dt], o[dt]);
    }
    __syncthreads();
  }

  // normalize + write upd[b][q][h*64+d] (head-major flatten) as bf16
#pragma unroll
  for (int dt = 0; dt < 4; ++dt)
#pragma unroll
    for (int vr = 0; vr < 8; ++vr) {
      int qrow = qw + vr + 8 * kh;
      int d = dt * 16 + r;
      float val = o[dt][vr] / lrow[vr];
      upd[(size_t)(b * 512 + qrow) * 512 + h * 64 + d] = (__bf16)val;
    }
}

// ---------------------------------------------------------------------------
// Fused 3-layer MLP per 64-row tile (4 waves):
//   h1 = silu(U@W1+b1) [64x128] -> LDS, h2 = silu(h1@W2+b2) [64x64] -> LDS,
//   out = silu(h2@W3+b3) [64x256] -> global f32
// ---------------------------------------------------------------------------
__global__ __launch_bounds__(128) void k_mlp(const __bf16* __restrict__ U,
                                             const __bf16* __restrict__ W1,
                                             const float* __restrict__ b1,
                                             const __bf16* __restrict__ W2,
                                             const float* __restrict__ b2,
                                             const __bf16* __restrict__ W3,
                                             const float* __restrict__ b3,
                                             float* __restrict__ out) {
  __shared__ __align__(16) char smem[53248];
  __bf16* As  = (__bf16*)smem;             // 4 KB  [64][32]      (phase 1)
  __bf16* Bt1 = (__bf16*)(smem + 4096);    // 8 KB  [128][32]     (phase 1)
  __bf16* Bt2 = (__bf16*)(smem + 4096);    // 4 KB  [64][32]      (phase 2)
  __bf16* h1  = (__bf16*)(smem + 12288);   // 16 KB [64][128]     (p1 out, p2 in)
  __bf16* h2  = (__bf16*)(smem + 45056);   // 8 KB  [64][64]      (p2 out, p3 in)
  __bf16* Bt3 = (__bf16*)smem;             // 32 KB [256][64]     (phase 3)

  const int tid = threadIdx.x;
  const int wave = tid >> 5, lane = tid & 31;
  const int r = lane & 15, kh = lane >> 4;
  const int m0 = blockIdx.x * 64;

  // ---- phase 1: K = 512
  v8f acc1[8];
#pragma unroll
  for (int i = 0; i < 8; ++i) acc1[i] = 0.0f;
  for (int k0 = 0; k0 < 512; k0 += 32) {
    { // A tile via async global->LDS
      int row = tid >> 1, col = (tid & 1) * 16;
      const __bf16* src = &U[(size_t)(m0 + row) * 512 + k0 + col];
      async_copy_b128(&As[row * 32 + col], src);
      async_copy_b128(&As[row * 32 + col + 8], src + 8);
    }
#pragma unroll
    for (int i = 0; i < 4; ++i) {
      int g = tid + 128 * i;                  // 0..511 groups of 8
      int k = g >> 4, n8 = (g & 15) * 8;
      v8bf w = *(const v8bf*)&W1[(size_t)(k0 + k) * 128 + n8];
#pragma unroll
      for (int j = 0; j < 8; ++j) Bt1[(n8 + j) * 32 + k] = w[j];
    }
    async_wait0();
    __syncthreads();
    v16bf a = frag_a(&As[(wave * 16 + r) * 32], 0, kh);
#pragma unroll
    for (int g = 0; g < 8; g += 4) {
      v16bf bf4[4];
#pragma unroll
      for (int nt = 0; nt < 4; ++nt)
        bf4[nt] = frag_b(&Bt1[((g + nt) * 16 + r) * 32], 0, kh);
#pragma unroll
      for (int nt = 0; nt < 4; ++nt)
        acc1[g + nt] = wmma_bf16(a, bf4[nt], acc1[g + nt]);
    }
    __syncthreads();
  }
#pragma unroll
  for (int nt = 0; nt < 8; ++nt)
#pragma unroll
    for (int vr = 0; vr < 8; ++vr) {
      int col = nt * 16 + r;
      h1[(wave * 16 + vr + 8 * kh) * 128 + col] = (__bf16)silu_f(acc1[nt][vr] + b1[col]);
    }
  __syncthreads();

  // ---- phase 2: K = 128
  v8f acc2[4];
#pragma unroll
  for (int i = 0; i < 4; ++i) acc2[i] = 0.0f;
  for (int k0 = 0; k0 < 128; k0 += 32) {
#pragma unroll
    for (int i = 0; i < 2; ++i) {
      int g = tid + 128 * i;                  // 0..255 groups of 8
      int k = g >> 3, n8 = (g & 7) * 8;
      v8bf w = *(const v8bf*)&W2[(size_t)(k0 + k) * 64 + n8];
#pragma unroll
      for (int j = 0; j < 8; ++j) Bt2[(n8 + j) * 32 + k] = w[j];
    }
    __syncthreads();
    v16bf a = frag_a(&h1[(wave * 16 + r) * 128], k0, kh);
    v16bf bf4[4];
#pragma unroll
    for (int nt = 0; nt < 4; ++nt)
      bf4[nt] = frag_b(&Bt2[(nt * 16 + r) * 32], 0, kh);
#pragma unroll
    for (int nt = 0; nt < 4; ++nt)
      acc2[nt] = wmma_bf16(a, bf4[nt], acc2[nt]);
    __syncthreads();
  }
#pragma unroll
  for (int nt = 0; nt < 4; ++nt)
#pragma unroll
    for (int vr = 0; vr < 8; ++vr) {
      int col = nt * 16 + r;
      h2[(wave * 16 + vr + 8 * kh) * 64 + col] = (__bf16)silu_f(acc2[nt][vr] + b2[col]);
    }
  __syncthreads();

  // ---- phase 3: K = 64, N = 256 (two passes of 128)
#pragma unroll
  for (int i = 0; i < 16; ++i) {
    int g = tid + 128 * i;                    // 0..2047 groups of 8
    int k = g >> 5, n8 = (g & 31) * 8;
    v8bf w = *(const v8bf*)&W3[(size_t)k * 256 + n8];
#pragma unroll
    for (int j = 0; j < 8; ++j) Bt3[(n8 + j) * 64 + k] = w[j];
  }
  __syncthreads();
  for (int np = 0; np < 2; ++np) {
    v8f acc3[8];
#pragma unroll
    for (int i = 0; i < 8; ++i) acc3[i] = 0.0f;
#pragma unroll
    for (int kk = 0; kk < 2; ++kk) {
      v16bf a = frag_a(&h2[(wave * 16 + r) * 64], kk * 32, kh);
#pragma unroll
      for (int g = 0; g < 8; g += 4) {
        v16bf bf4[4];
#pragma unroll
        for (int nt = 0; nt < 4; ++nt)
          bf4[nt] = frag_b(&Bt3[(np * 128 + (g + nt) * 16 + r) * 64], kk * 32, kh);
#pragma unroll
        for (int nt = 0; nt < 4; ++nt)
          acc3[g + nt] = wmma_bf16(a, bf4[nt], acc3[g + nt]);
      }
    }
#pragma unroll
    for (int nt = 0; nt < 8; ++nt)
#pragma unroll
      for (int vr = 0; vr < 8; ++vr) {
        int col = np * 128 + nt * 16 + r;
        out[(size_t)(m0 + wave * 16 + vr + 8 * kh) * 256 + col] =
            silu_f(acc3[nt][vr] + b3[col]);
      }
  }
}

// ---------------------------------------------------------------------------
extern "C" void kernel_launch(void* const* d_in, const int* in_sizes, int n_in,
                              void* d_out, int out_size, void* d_ws, size_t ws_size,
                              hipStream_t stream) {
  (void)in_sizes; (void)n_in; (void)out_size; (void)ws_size;
  const float* kp_pos  = (const float*)d_in[0];
  const float* kp_feat = (const float*)d_in[1];
  const float* Wk = (const float*)d_in[2];
  const float* Wq = (const float*)d_in[3];
  const float* Wv = (const float*)d_in[4];
  const float* W1 = (const float*)d_in[5];
  const float* b1 = (const float*)d_in[6];
  const float* W2 = (const float*)d_in[7];
  const float* b2 = (const float*)d_in[8];
  const float* W3 = (const float*)d_in[9];
  const float* b3 = (const float*)d_in[10];
  float* out = (float*)d_out;

  __bf16* ws = (__bf16*)d_ws;
  size_t o = 0;
  __bf16* Xb  = ws + o; o += 16384ull * 256;   // kp_feat bf16
  __bf16* Wkb = ws + o; o += 256 * 512;
  __bf16* Wqb = ws + o; o += 256 * 512;
  __bf16* Wvb = ws + o; o += 256 * 512;
  __bf16* W1b = ws + o; o += 512 * 128;
  __bf16* W2b = ws + o; o += 128 * 64;
  __bf16* W3b = ws + o; o += 64 * 256;
  __bf16* Qb  = ws + o; o += 16384ull * 512;   // [B,H,N,64]
  __bf16* Kb  = ws + o; o += 16384ull * 512;
  __bf16* Vb  = ws + o; o += 16384ull * 512;
  __bf16* Ub  = ws + o; o += 16384ull * 512;   // upd [16384,512]

  k_convert<<<16384, 256, 0, stream>>>(kp_feat, Xb, 16384 * 256);
  k_convert<<<512, 256, 0, stream>>>(Wk, Wkb, 256 * 512);
  k_convert<<<512, 256, 0, stream>>>(Wq, Wqb, 256 * 512);
  k_convert<<<512, 256, 0, stream>>>(Wv, Wvb, 256 * 512);
  k_convert<<<256, 256, 0, stream>>>(W1, W1b, 512 * 128);
  k_convert<<<32, 256, 0, stream>>>(W2, W2b, 128 * 64);
  k_convert<<<64, 256, 0, stream>>>(W3, W3b, 64 * 256);

  k_gemm_qkv<<<dim3(128, 8), 256, 0, stream>>>(Xb, Wqb, Qb);
  k_gemm_qkv<<<dim3(128, 8), 256, 0, stream>>>(Xb, Wkb, Kb);
  k_gemm_qkv<<<dim3(128, 8), 256, 0, stream>>>(Xb, Wvb, Vb);

  k_attn<<<dim3(4, 8, 32), 256, 0, stream>>>(kp_pos, Qb, Kb, Vb, Ub);

  k_mlp<<<256, 128, 0, stream>>>(Ub, W1b, b1, W2b, b2, W3b, b3, out);
}
